// FastSpeechModel_44899588112698
// MI455X (gfx1250) — matrix operand order repaired and checked by reading, not compiled, not run on hardware
//
#include <hip/hip_runtime.h>
#include <hip/hip_bf16.h>

typedef __attribute__((ext_vector_type(16))) _Float16 v16h;
typedef __attribute__((ext_vector_type(8)))  _Float16 v8h;
typedef __attribute__((ext_vector_type(8)))  float    v8f;

#define NBATCH 8
#define TTOK   128
#define DMODEL 384
#define NHEAD  6
#define DHEAD  64
#define NLAYER 4
#define DINTER 1536
#define TMELX  1024
#define NOUT   80

// ---------------------------------------------------------------- WMMA helpers
__device__ __forceinline__ v8f wmma16(v16h a, v16h b, v8f c) {
  return __builtin_amdgcn_wmma_f32_16x16x32_f16(false, a, false, b, (short)0, c, false, false);
}

__device__ __forceinline__ v16h join8(v8h lo, v8h hi) {
  return __builtin_shufflevector(lo, hi, 0, 1, 2, 3, 4, 5, 6, 7, 8, 9, 10, 11, 12, 13, 14, 15);
}

// A fragment (16x32) from f16 row-major [rows x ldA]:
// lane L (m=L%16, half=L/16), e -> A[rowbase+m][kb + (e/8)*16 + half*8 + e%8]
__device__ __forceinline__ v16h load_a16(const _Float16* __restrict__ A, long ldA,
                                         long rowbase, long kb, int lane) {
  int m = lane & 15, half = (lane >> 4) & 1;
  const _Float16* p = A + (rowbase + m) * ldA + kb + half * 8;
  v8h lo = *(const v8h*)p;
  v8h hi = *(const v8h*)(p + 16);
  return join8(lo, hi);
}

// B fragment (32x16) from f16 [Nrows x ldW] with K contiguous per row (W transposed):
// lane L (n=L%16, half=L/16), e -> W[(ncol+n)*ldW + kb + (e/8)*16 + half*8 + e%8]
__device__ __forceinline__ v16h load_bT16(const _Float16* __restrict__ W, long ldW,
                                          long ncol, long kb, int lane) {
  int n = lane & 15, half = (lane >> 4) & 1;
  const _Float16* p = W + (ncol + n) * ldW + kb + half * 8;
  v8h lo = *(const v8h*)p;
  v8h hi = *(const v8h*)(p + 16);
  return join8(lo, hi);
}

// ---------------------------------------------------------------- weight transpose+convert: W[K,N](f32) -> WT[N,K](f16), per slice
__global__ void wconv_kernel(const float* __restrict__ src, _Float16* __restrict__ dst,
                             int Kd, int N, int cnt) {
  long i = (long)blockIdx.x * 256 + threadIdx.x;
  long per = (long)Kd * N;
  long tot = per * cnt;
  if (i >= tot) return;
  long s = i / per, rem = i - s * per;
  long k = rem / N, n = rem - k * N;
  dst[s * per + n * Kd + k] = (_Float16)src[i];
}

// ---------------------------------------------------------------- embedding + PE (f32 + f16 shadow)
__global__ void embed_pe_kernel(const int* __restrict__ tokens, const float* __restrict__ emb,
                                float* __restrict__ X, _Float16* __restrict__ Xh) {
  long i = (long)blockIdx.x * blockDim.x + threadIdx.x;
  const long total = (long)NBATCH * TTOK * DMODEL;
  if (i >= total) return;
  int d = (int)(i % DMODEL);
  long bt = i / DMODEL;
  int b = (int)(bt / TTOK);
  int tok = tokens[bt];
  // faithful quirk: PE indexed by BATCH index, broadcast over time; x = 2*emb + pe[b]
  float den = __expf(-(float)(d & ~1) * (9.210340371976184f / (float)DMODEL));
  float ang = (float)b * den;
  float pe = (d & 1) ? cosf(ang) : sinf(ang);
  float v = 2.0f * emb[(long)tok * DMODEL + d] + pe;
  X[i] = v;
  Xh[i] = (_Float16)v;
}

// ---------------------------------------------------------------- generic WMMA GEMM (f16 in, f32/f16 out), K-pipelined
// C = A[M,Kdim] @ WT^T (+bias)(+relu); WT is [N,Kdim] f16. grid (M/64, N/64), block 128.
// Kdim must be a multiple of 32 here (384/1152/4608 all are) -> loop stays pipelined.
__global__ void gemm_kernel(const _Float16* __restrict__ A, const _Float16* __restrict__ WT,
                            const float* __restrict__ bias, float* __restrict__ Cf,
                            _Float16* __restrict__ Ch, int Kdim, int N, int relu,
                            int vtrans, int Tloc) {
  int lane = threadIdx.x & 31, wave = threadIdx.x >> 5;
  long rowbase = ((long)blockIdx.x * 4 + wave) * 16;
  long colbase = (long)blockIdx.y * 64;
  v8f acc[4] = {};

  // software pipeline: fragments for step i+1 are in flight while WMMAs of step i issue
  v16h a0 = load_a16(A, Kdim, rowbase, 0, lane);
  v16h b0[4];
#pragma unroll
  for (int nt = 0; nt < 4; ++nt) b0[nt] = load_bT16(WT, Kdim, colbase + nt * 16, 0, lane);

  int kb = 0;
  for (; kb + 32 < Kdim; kb += 32) {
    v16h a1 = load_a16(A, Kdim, rowbase, kb + 32, lane);
    v16h b1[4];
#pragma unroll
    for (int nt = 0; nt < 4; ++nt) b1[nt] = load_bT16(WT, Kdim, colbase + nt * 16, kb + 32, lane);
#pragma unroll
    for (int nt = 0; nt < 4; ++nt) acc[nt] = wmma16(a0, b0[nt], acc[nt]);
    a0 = a1;
#pragma unroll
    for (int nt = 0; nt < 4; ++nt) b0[nt] = b1[nt];
  }
#pragma unroll
  for (int nt = 0; nt < 4; ++nt) acc[nt] = wmma16(a0, b0[nt], acc[nt]);

  int nloc = lane & 15, half = lane >> 4;
#pragma unroll
  for (int nt = 0; nt < 4; ++nt) {
    long col = colbase + nt * 16 + nloc;
    float bv = bias ? bias[col] : 0.0f;
#pragma unroll
    for (int r = 0; r < 8; ++r) {
      float v = acc[nt][r] + bv;
      if (relu) v = fmaxf(v, 0.0f);
      long row = rowbase + half * 8 + r;
      if (Cf) Cf[row * (long)N + col] = v;
      if (Ch) {
        if (vtrans) {
          int hh = (int)(col >> 6), dh = (int)(col & 63);
          long bb = row / Tloc;
          long t = row - bb * Tloc;
          Ch[((bb * NHEAD + hh) * DHEAD + dh) * (long)Tloc + t] = (_Float16)v;
        } else {
          Ch[row * (long)N + col] = (_Float16)v;
        }
      }
    }
  }
}

// ---------------------------------------------------------------- conv1d K=3 SAME as implicit-im2col GEMM (f16 in), K-pipelined
__global__ void convgemm_kernel(const _Float16* __restrict__ X, const _Float16* __restrict__ WT,
                                const float* __restrict__ bias, float* __restrict__ Cf,
                                _Float16* __restrict__ Ch, int Tloc, int Cin, int Cout,
                                int relu) {
  int lane = threadIdx.x & 31, wave = threadIdx.x >> 5;
  long rowbase = ((long)blockIdx.x * 4 + wave) * 16;
  long colbase = (long)blockIdx.y * 64;
  int m = lane & 15, half = lane >> 4;
  long row = rowbase + m;
  long bb = row / Tloc;
  int t = (int)(row - bb * Tloc);
  const int Kdim = 3 * Cin;

  // A-fragment gather: two 8-half runs, each inside one kernel tap (Cin % 8 == 0)
  auto load_a_conv = [&](int kbb) -> v16h {
    v8h parts[2];
#pragma unroll
    for (int g = 0; g < 2; ++g) {
      int c0 = kbb + g * 16 + half * 8;
      int dk = c0 / Cin;
      int ci = c0 - dk * Cin;
      int tt = t + dk - 1;
      v8h z = {};
      parts[g] = (tt >= 0 && tt < Tloc)
                     ? *(const v8h*)(X + (bb * Tloc + tt) * (long)Cin + ci)
                     : z;
    }
    return join8(parts[0], parts[1]);
  };

  v8f acc[4] = {};
  v16h a0 = load_a_conv(0);
  v16h b0[4];
#pragma unroll
  for (int nt = 0; nt < 4; ++nt) b0[nt] = load_bT16(WT, Kdim, colbase + nt * 16, 0, lane);

  int kb = 0;
  for (; kb + 32 < Kdim; kb += 32) {
    v16h a1 = load_a_conv(kb + 32);
    v16h b1[4];
#pragma unroll
    for (int nt = 0; nt < 4; ++nt) b1[nt] = load_bT16(WT, Kdim, colbase + nt * 16, kb + 32, lane);
#pragma unroll
    for (int nt = 0; nt < 4; ++nt) acc[nt] = wmma16(a0, b0[nt], acc[nt]);
    a0 = a1;
#pragma unroll
    for (int nt = 0; nt < 4; ++nt) b0[nt] = b1[nt];
  }
#pragma unroll
  for (int nt = 0; nt < 4; ++nt) acc[nt] = wmma16(a0, b0[nt], acc[nt]);

  int nloc = lane & 15;
#pragma unroll
  for (int nt = 0; nt < 4; ++nt) {
    long col = colbase + nt * 16 + nloc;
    float bv = bias[col];
#pragma unroll
    for (int r = 0; r < 8; ++r) {
      float v = acc[nt][r] + bv;
      if (relu) v = fmaxf(v, 0.0f);
      long orow = rowbase + half * 8 + r;
      if (Cf) Cf[orow * (long)Cout + col] = v;
      if (Ch) Ch[orow * (long)Cout + col] = (_Float16)v;
    }
  }
}

// ---------------------------------------------------------------- flash attention (wave per 16-query tile), f16 I/O
// Q,K: [B,Tloc,NHEAD,DHEAD] f16; Vt: [B,NHEAD,DHEAD,Tloc] f16; O: [B,Tloc,NHEAD,DHEAD] f16.
__global__ void attn_kernel(const _Float16* __restrict__ Q, const _Float16* __restrict__ K,
                            const _Float16* __restrict__ Vt, _Float16* __restrict__ O,
                            const int* __restrict__ lens, int Tloc) {
  __shared__ _Float16 pbuf[4][16 * 32];
  int lane = threadIdx.x & 31, wave = threadIdx.x >> 5;
  int b = blockIdx.x / NHEAD, h = blockIdx.x % NHEAD;
  long qbase = ((long)blockIdx.y * 4 + wave) * 16;
  const long ld = NHEAD * DHEAD;
  const _Float16* Qp = Q + (long)b * Tloc * ld + h * DHEAD;
  const _Float16* Kp = K + (long)b * Tloc * ld + h * DHEAD;
  const _Float16* Vp = Vt + ((long)b * NHEAD + h) * DHEAD * (long)Tloc;
  _Float16* Op = O + (long)b * Tloc * ld + h * DHEAD;
  int len = lens[b];
  int nloc = lane & 15, half = lane >> 4;

  v16h aq0 = load_a16(Qp, ld, qbase, 0, lane);
  v16h aq1 = load_a16(Qp, ld, qbase, 32, lane);

  float mrow[8], lrow[8];
  v8f acc[4] = {};
#pragma unroll
  for (int r = 0; r < 8; ++r) { mrow[r] = -3.0e38f; lrow[r] = 0.0f; }
  _Float16* pl = pbuf[wave];

  for (int kb = 0; kb < Tloc; kb += 32) {
    v8f s0 = {}, s1 = {};
    {
      v16h bk = load_bT16(Kp, ld, kb, 0, lane);
      s0 = wmma16(aq0, bk, s0);
      bk = load_bT16(Kp, ld, kb, 32, lane);
      s0 = wmma16(aq1, bk, s0);
      bk = load_bT16(Kp, ld, kb + 16, 0, lane);
      s1 = wmma16(aq0, bk, s1);
      bk = load_bT16(Kp, ld, kb + 16, 32, lane);
      s1 = wmma16(aq1, bk, s1);
    }
    bool msk0 = (kb + nloc) > len;
    bool msk1 = (kb + 16 + nloc) > len;
#pragma unroll
    for (int r = 0; r < 8; ++r) {
      float t0 = msk0 ? -1.0e30f : s0[r] * 0.125f;   // 1/sqrt(64)
      float t1 = msk1 ? -1.0e30f : s1[r] * 0.125f;
      float mt = fmaxf(t0, t1);
#pragma unroll
      for (int xm = 1; xm < 16; xm <<= 1) mt = fmaxf(mt, __shfl_xor(mt, xm, 32));
      float nm = fmaxf(mrow[r], mt);
      float sc = __expf(mrow[r] - nm);
      mrow[r] = nm;
      float p0 = __expf(t0 - nm);
      float p1 = __expf(t1 - nm);
      float rs = p0 + p1;
#pragma unroll
      for (int xm = 1; xm < 16; xm <<= 1) rs += __shfl_xor(rs, xm, 32);
      lrow[r] = lrow[r] * sc + rs;
      acc[0][r] *= sc; acc[1][r] *= sc; acc[2][r] *= sc; acc[3][r] *= sc;
      pl[(half * 8 + r) * 32 + nloc] = (_Float16)p0;
      pl[(half * 8 + r) * 32 + 16 + nloc] = (_Float16)p1;
    }
    asm volatile("" ::: "memory");   // order DS writes -> DS reads (same wave; LDS is in-order)
    const _Float16* pr = pl + nloc * 32 + half * 8;
    v16h pa = join8(*(const v8h*)pr, *(const v8h*)(pr + 16));
    asm volatile("" ::: "memory");
#pragma unroll
    for (int nt = 0; nt < 4; ++nt) {
      v16h bv = load_bT16(Vp, Tloc, nt * 16, kb, lane);  // Vt rows = dh, contiguous over keys
      acc[nt] = wmma16(pa, bv, acc[nt]);
    }
  }
#pragma unroll
  for (int nt = 0; nt < 4; ++nt) {
#pragma unroll
    for (int r = 0; r < 8; ++r) {
      Op[(qbase + half * 8 + r) * ld + nt * 16 + nloc] = (_Float16)(acc[nt][r] / lrow[r]);
    }
  }
}

// ---------------------------------------------------------------- LayerNorm (optional residual), f32 out + optional f16 shadow
__global__ void ln_kernel(const float* __restrict__ X, const float* __restrict__ R,
                          const float* __restrict__ g, const float* __restrict__ bt,
                          float* __restrict__ Y, _Float16* __restrict__ Yh) {
  int lane = threadIdx.x & 31, wave = threadIdx.x >> 5;
  long row = (long)blockIdx.x * 8 + wave;
  const float* xr = X + row * DMODEL;
  const float* rr = R ? (R + row * DMODEL) : nullptr;
  float v[12];
  float s = 0.0f;
#pragma unroll
  for (int j = 0; j < 12; ++j) {
    int c = lane + 32 * j;
    float t = xr[c] + (rr ? rr[c] : 0.0f);
    v[j] = t; s += t;
  }
#pragma unroll
  for (int xm = 1; xm < 32; xm <<= 1) s += __shfl_xor(s, xm, 32);
  float mean = s * (1.0f / DMODEL);
  float s2 = 0.0f;
#pragma unroll
  for (int j = 0; j < 12; ++j) { float d = v[j] - mean; s2 += d * d; }
#pragma unroll
  for (int xm = 1; xm < 32; xm <<= 1) s2 += __shfl_xor(s2, xm, 32);
  float inv = rsqrtf(s2 * (1.0f / DMODEL) + 1e-5f);
#pragma unroll
  for (int j = 0; j < 12; ++j) {
    int c = lane + 32 * j;
    float o = (v[j] - mean) * inv * g[c] + bt[c];
    Y[row * DMODEL + c] = o;
    if (Yh) Yh[row * DMODEL + c] = (_Float16)o;
  }
}

// ---------------------------------------------------------------- duration cumsum + mel length
__global__ void cumsum_kernel(const int* __restrict__ dur, int* __restrict__ cs,
                              int* __restrict__ mel) {
  int b = threadIdx.x;
  if (b < NBATCH) {
    int s = 0;
    for (int t = 0; t < TTOK; ++t) { s += dur[b * TTOK + t]; cs[b * TTOK + t] = s; }
    mel[b] = s;
  }
}

// ---------------------------------------------------------------- length regulator (wave per mel position)
__global__ void lr_kernel(const float* __restrict__ enc, const int* __restrict__ cs,
                          const int* __restrict__ mel, float* __restrict__ Xd,
                          _Float16* __restrict__ Xdh) {
  int lane = threadIdx.x & 31, wave = threadIdx.x >> 5;
  long idx = (long)blockIdx.x * 8 + wave;        // b*TMEL + m
  int b = (int)(idx / TMELX);
  int m = (int)(idx % TMELX);
  int ml = mel[b];
  if (m < ml) {
    const int* c = cs + b * TTOK;
    int t = 0;
    while (t < TTOK - 1 && c[t] <= m) ++t;       // first t with cs[t] > m
    const float* er = enc + ((long)b * TTOK + t) * DMODEL;
#pragma unroll
    for (int j = 0; j < 12; ++j) {
      float v = er[lane + 32 * j];
      Xd[idx * DMODEL + lane + 32 * j] = v;
      Xdh[idx * DMODEL + lane + 32 * j] = (_Float16)v;
    }
  } else {
#pragma unroll
    for (int j = 0; j < 12; ++j) {
      Xd[idx * DMODEL + lane + 32 * j] = 0.0f;
      Xdh[idx * DMODEL + lane + 32 * j] = (_Float16)0.0f;
    }
  }
}

// ---------------------------------------------------------------- duration predictor head (384 -> 1)
__global__ void dp_final_kernel(const float* __restrict__ Hx, const float* __restrict__ w,
                                const float* __restrict__ b0, float* __restrict__ lp) {
  int lane = threadIdx.x & 31, wave = threadIdx.x >> 5;
  long row = (long)blockIdx.x * 8 + wave;
  float s = 0.0f;
#pragma unroll
  for (int j = 0; j < 12; ++j) s += Hx[row * DMODEL + lane + 32 * j] * w[lane + 32 * j];
#pragma unroll
  for (int xm = 1; xm < 32; xm <<= 1) s += __shfl_xor(s, xm, 32);
  if (lane == 0) __builtin_nontemporal_store(s + b0[0], lp + row);
}

// ---------------------------------------------------------------- final projection, transposed NT store [B,OUT,TMEL] (f32)
__global__ void outproj_kernel(const _Float16* __restrict__ A, const _Float16* __restrict__ WT,
                               const float* __restrict__ bias, float* __restrict__ Out) {
  int lane = threadIdx.x & 31, wave = threadIdx.x >> 5;
  long rowbase = ((long)blockIdx.x * 4 + wave) * 16;
  v8f acc[5] = {};

  v16h a0 = load_a16(A, DMODEL, rowbase, 0, lane);
  v16h b0[5];
#pragma unroll
  for (int nt = 0; nt < 5; ++nt) b0[nt] = load_bT16(WT, DMODEL, nt * 16, 0, lane);
  int kb = 0;
  for (; kb + 32 < DMODEL; kb += 32) {
    v16h a1 = load_a16(A, DMODEL, rowbase, kb + 32, lane);
    v16h b1[5];
#pragma unroll
    for (int nt = 0; nt < 5; ++nt) b1[nt] = load_bT16(WT, DMODEL, nt * 16, kb + 32, lane);
#pragma unroll
    for (int nt = 0; nt < 5; ++nt) acc[nt] = wmma16(a0, b0[nt], acc[nt]);
    a0 = a1;
#pragma unroll
    for (int nt = 0; nt < 5; ++nt) b0[nt] = b1[nt];
  }
#pragma unroll
  for (int nt = 0; nt < 5; ++nt) acc[nt] = wmma16(a0, b0[nt], acc[nt]);

  int nloc = lane & 15, half = lane >> 4;
#pragma unroll
  for (int nt = 0; nt < 5; ++nt) {
    int col = nt * 16 + nloc;
    float bv = bias[col];
#pragma unroll
    for (int r = 0; r < 8; ++r) {
      long row = rowbase + half * 8 + r;
      int bb = (int)(row >> 10);
      int mm = (int)(row & 1023);
      __builtin_nontemporal_store(acc[nt][r] + bv,
                                  Out + (long)bb * NOUT * TMELX + (long)col * TMELX + mm);
    }
  }
}

// ================================================================ host driver
extern "C" void kernel_launch(void* const* d_in, const int* in_sizes, int n_in,
                              void* d_out, int out_size, void* d_ws, size_t ws_size,
                              hipStream_t stream) {
  (void)in_sizes; (void)n_in; (void)out_size; (void)ws_size;
  const int*   tokens = (const int*)d_in[0];
  const int*   toklen = (const int*)d_in[1];
  const int*   dur    = (const int*)d_in[2];
  const float* emb    = (const float*)d_in[3];

  const float* wq[2]   = {(const float*)d_in[4],  (const float*)d_in[16]};
  const float* wk[2]   = {(const float*)d_in[5],  (const float*)d_in[17]};
  const float* wv[2]   = {(const float*)d_in[6],  (const float*)d_in[18]};
  const float* wo[2]   = {(const float*)d_in[7],  (const float*)d_in[19]};
  const float* ln1g[2] = {(const float*)d_in[8],  (const float*)d_in[20]};
  const float* ln1b[2] = {(const float*)d_in[9],  (const float*)d_in[21]};
  const float* c1w[2]  = {(const float*)d_in[10], (const float*)d_in[22]};
  const float* c1b[2]  = {(const float*)d_in[11], (const float*)d_in[23]};
  const float* c2w[2]  = {(const float*)d_in[12], (const float*)d_in[24]};
  const float* c2b[2]  = {(const float*)d_in[13], (const float*)d_in[25]};
  const float* ln2g[2] = {(const float*)d_in[14], (const float*)d_in[26]};
  const float* ln2b[2] = {(const float*)d_in[15], (const float*)d_in[27]};

  const float* dp_c1w  = (const float*)d_in[28];
  const float* dp_c1b  = (const float*)d_in[29];
  const float* dp_ln1g = (const float*)d_in[30];
  const float* dp_ln1b = (const float*)d_in[31];
  const float* dp_c2w  = (const float*)d_in[32];
  const float* dp_c2b  = (const float*)d_in[33];
  const float* dp_ln2g = (const float*)d_in[34];
  const float* dp_ln2b = (const float*)d_in[35];
  const float* dp_w    = (const float*)d_in[36];
  const float* dp_b    = (const float*)d_in[37];
  const float* out_w   = (const float*)d_in[38];
  const float* out_b   = (const float*)d_in[39];

  float* out = (float*)d_out;
  float* lp  = out + (long)NBATCH * NOUT * TMELX;   // length_predictions tail

  // ---- workspace carve (16B aligned chunks)
  char* wsb = (char*)d_ws;
  auto allocF = [&](long n) { float* p = (float*)wsb; wsb += ((n * 4 + 15) & ~15L); return p; };
  auto allocH = [&](long n) { _Float16* p = (_Float16*)wsb; wsb += ((n * 2 + 15) & ~15L); return p; };
  auto allocI = [&](long n) { int* p = (int*)wsb; wsb += ((n * 4 + 15) & ~15L); return p; };

  const long S = (long)8192 * DMODEL;
  float* XA = allocF(S);           // activations A (f32)
  float* XB = allocF(S);           // activations B (f32)
  float* O2 = allocF(S);           // proj/conv2 output (f32)
  int* cs   = allocI(NBATCH * TTOK);
  int* mel  = allocI(16);

  _Float16* XAh = allocH(S);
  _Float16* XBh = allocH(S);
  _Float16* Qh  = allocH(S);
  _Float16* Kh  = allocH(S);
  _Float16* Obh = allocH(S);                   // attention output (f16)
  _Float16* VtH = allocH(S);                   // V transposed [B,H,DH,T] (f16)
  _Float16* Hbh = allocH((long)8192 * DINTER); // conv1 intermediate (f16)

  _Float16* wqT[2], *wkT[2], *wvT[2], *woT[2], *c1T[2], *c2T[2];
  for (int si = 0; si < 2; ++si) {
    wqT[si] = allocH((long)NLAYER * DMODEL * DMODEL);
    wkT[si] = allocH((long)NLAYER * DMODEL * DMODEL);
    wvT[si] = allocH((long)NLAYER * DMODEL * DMODEL);
    woT[si] = allocH((long)NLAYER * DMODEL * DMODEL);
    c1T[si] = allocH((long)NLAYER * 3 * DMODEL * DINTER);
    c2T[si] = allocH((long)NLAYER * 3 * DINTER * DMODEL);
  }
  _Float16* dp1T  = allocH((long)3 * DMODEL * DMODEL);
  _Float16* dp2T  = allocH((long)3 * DMODEL * DMODEL);
  _Float16* outwT = allocH((long)DMODEL * NOUT);

  auto wconv = [&](const float* src, _Float16* dst, long Kd, long N, long cnt) {
    long tot = Kd * N * cnt;
    wconv_kernel<<<dim3((unsigned)((tot + 255) / 256)), 256, 0, stream>>>(src, dst, (int)Kd, (int)N, (int)cnt);
  };

  // ---- one-time weight convert+transpose to f16 [N,K]
  for (int si = 0; si < 2; ++si) {
    wconv(wq[si], wqT[si], DMODEL, DMODEL, NLAYER);
    wconv(wk[si], wkT[si], DMODEL, DMODEL, NLAYER);
    wconv(wv[si], wvT[si], DMODEL, DMODEL, NLAYER);
    wconv(wo[si], woT[si], DMODEL, DMODEL, NLAYER);
    wconv(c1w[si], c1T[si], 3 * DMODEL, DINTER, NLAYER);
    wconv(c2w[si], c2T[si], 3 * DINTER, DMODEL, NLAYER);
  }
  wconv(dp_c1w, dp1T, 3 * DMODEL, DMODEL, 1);
  wconv(dp_c2w, dp2T, 3 * DMODEL, DMODEL, 1);
  wconv(out_w, outwT, DMODEL, NOUT, 1);

  // ---- embedding + PE, duration prefix sums
  embed_pe_kernel<<<dim3((NBATCH * TTOK * DMODEL + 255) / 256), 256, 0, stream>>>(tokens, emb, XA, XAh);
  cumsum_kernel<<<dim3(1), 32, 0, stream>>>(dur, cs, mel);

  auto run_stack = [&](int si, float* X, _Float16* Xh, float* XT, _Float16* XTh,
                       int M, int Tloc, const int* lens) {
    dim3 gproj(M / 64, DMODEL / 64);
    for (int l = 0; l < NLAYER; ++l) {
      const _Float16* wql = wqT[si] + (long)l * DMODEL * DMODEL;
      const _Float16* wkl = wkT[si] + (long)l * DMODEL * DMODEL;
      const _Float16* wvl = wvT[si] + (long)l * DMODEL * DMODEL;
      const _Float16* wol = woT[si] + (long)l * DMODEL * DMODEL;
      gemm_kernel<<<gproj, 128, 0, stream>>>(Xh, wql, nullptr, nullptr, Qh, DMODEL, DMODEL, 0, 0, Tloc);
      gemm_kernel<<<gproj, 128, 0, stream>>>(Xh, wkl, nullptr, nullptr, Kh, DMODEL, DMODEL, 0, 0, Tloc);
      gemm_kernel<<<gproj, 128, 0, stream>>>(Xh, wvl, nullptr, nullptr, VtH, DMODEL, DMODEL, 0, 1, Tloc);
      attn_kernel<<<dim3(NBATCH * NHEAD, Tloc / 64), 128, 0, stream>>>(Qh, Kh, VtH, Obh, lens, Tloc);
      gemm_kernel<<<gproj, 128, 0, stream>>>(Obh, wol, nullptr, O2, nullptr, DMODEL, DMODEL, 0, 0, Tloc);
      ln_kernel<<<dim3(M / 8), 256, 0, stream>>>(X, O2, ln1g[si] + l * DMODEL, ln1b[si] + l * DMODEL, XT, XTh);
      convgemm_kernel<<<dim3(M / 64, DINTER / 64), 128, 0, stream>>>(
          XTh, c1T[si] + (long)l * 3 * DMODEL * DINTER, c1b[si] + l * DINTER, nullptr, Hbh,
          Tloc, DMODEL, DINTER, 1);
      convgemm_kernel<<<dim3(M / 64, DMODEL / 64), 128, 0, stream>>>(
          Hbh, c2T[si] + (long)l * 3 * DINTER * DMODEL, c2b[si] + l * DMODEL, O2, nullptr,
          Tloc, DINTER, DMODEL, 0);
      ln_kernel<<<dim3(M / 8), 256, 0, stream>>>(XT, O2, ln2g[si] + l * DMODEL, ln2b[si] + l * DMODEL, X, Xh);
    }
  };

  // ---- encoder (result in XA / XAh)
  run_stack(0, XA, XAh, XB, XBh, NBATCH * TTOK, TTOK, toklen);

  // ---- duration predictor on encoder output (XB/XBh used as temps; LR rewrites them after)
  convgemm_kernel<<<dim3(16, 6), 128, 0, stream>>>(XAh, dp1T, dp_c1b, O2, nullptr, TTOK, DMODEL, DMODEL, 1);
  ln_kernel<<<dim3(128), 256, 0, stream>>>(O2, nullptr, dp_ln1g, dp_ln1b, XB, XBh);
  convgemm_kernel<<<dim3(16, 6), 128, 0, stream>>>(XBh, dp2T, dp_c2b, O2, nullptr, TTOK, DMODEL, DMODEL, 1);
  ln_kernel<<<dim3(128), 256, 0, stream>>>(O2, nullptr, dp_ln2g, dp_ln2b, XB, nullptr);
  dp_final_kernel<<<dim3(128), 256, 0, stream>>>(XB, dp_w, dp_b, lp);

  // ---- length regulator: XA (enc, f32) -> XB/XBh (decoder input)
  lr_kernel<<<dim3(NBATCH * TMELX / 8), 256, 0, stream>>>(XA, cs, mel, XB, XBh);

  // ---- decoder (result in XB / XBh)
  run_stack(1, XB, XBh, XA, XAh, NBATCH * TMELX, TMELX, mel);

  // ---- final projection -> [B, OUT, TMEL]
  outproj_kernel<<<dim3((NBATCH * TMELX / 16) / 4), 128, 0, stream>>>(XBh, outwT, out_b, out);
}
